// MoleculeTransformerGPT_21122649162323
// MI455X (gfx1250) — matrix operand order repaired and checked by reading, not compiled, not run on hardware
//
#include <hip/hip_runtime.h>
#include <math.h>

// ---------------------------------------------------------------------------
// MoleculeTransformerGPT block for MI455X (gfx1250, wave32, WMMA bf16/f32).
// All heavy matmuls use v_wmma_f32_16x16x32_bf16 (bf16 in, f32 accumulate).
// Each wave computes a 32x64 tile (8 WMMAs per k-step, B-fragments reused
// across two M-tiles). global_prefetch_b8 pulls next-next k-step A rows.
// Workspace requirement: ~230 MB (dominated by 128 MB bf16 attention tensor).
// ---------------------------------------------------------------------------

typedef __attribute__((ext_vector_type(16))) __bf16 v16bf;
typedef __attribute__((ext_vector_type(8)))  float  v8f;
typedef unsigned short u16;
typedef unsigned int   u32;

#define N_SEQ 2048
#define DIM   1024
#define HEADS 16
#define DHEAD 64

__device__ __forceinline__ u16 f2bf(float x) {
  u32 u = __float_as_uint(x);
  u32 r = u + 0x7FFFu + ((u >> 16) & 1u);   // round-to-nearest-even
  return (u16)(r >> 16);
}
__device__ __forceinline__ float bf2f(u16 b) {
  return __uint_as_float(((u32)b) << 16);
}

union ABFrag { v16bf v; uint4 q[2]; u16 us[16]; };

// A-fragment (16x32 bf16, MxK): lane<16 -> row=lane, K = {k0..k0+7, k0+16..k0+23}
//                               lane>=16 -> row=lane-16, K = {k0+8..15, k0+24..31}
__device__ __forceinline__ v16bf load_a_frag(const u16* A, int lda, int m0, int k0, int lane) {
  int row = m0 + (lane & 15);
  int kb  = k0 + ((lane & 16) ? 8 : 0);
  ABFrag f;
  const u16* p = A + (size_t)row * lda + kb;
  f.q[0] = *(const uint4*)(p);
  f.q[1] = *(const uint4*)(p + 16);
  return f.v;
}
// B-fragment (32x16 bf16, KxN) from a transposed (N-major, K-contiguous) buffer:
// lane<16 -> col=lane, K = {k0..k0+7, k0+16..k0+23} ; lane>=16 -> col=lane-16, K = {k0+8..15, k0+24..31}
__device__ __forceinline__ v16bf load_b_frag(const u16* Bt, int ldb, int n0, int k0, int lane) {
  int col = n0 + (lane & 15);
  int kb  = k0 + ((lane & 16) ? 8 : 0);
  ABFrag f;
  const u16* p = Bt + (size_t)col * ldb + kb;
  f.q[0] = *(const uint4*)(p);
  f.q[1] = *(const uint4*)(p + 16);
  return f.v;
}
__device__ __forceinline__ v8f wmma_bf16(v16bf a, v16bf b, v8f c) {
  return __builtin_amdgcn_wmma_f32_16x16x32_bf16(false, a, false, b, (short)0, c, false, false);
}
__device__ __forceinline__ float gelu_exact(float x) {
  return 0.5f * x * (1.0f + erff(x * 0.70710678118654752f));
}

// --------------------------- weight transpose+convert ----------------------
__global__ void k_cvtT(const float* __restrict__ W, u16* __restrict__ WT, int K, int Ncols) {
  int idx = blockIdx.x * 256 + threadIdx.x;
  if (idx >= K * Ncols) return;
  int k = idx / Ncols, n = idx % Ncols;
  WT[(size_t)n * K + k] = f2bf(W[idx]);
}

// --------------------------- LayerNorm (per row) ---------------------------
__global__ void k_ln(const float* __restrict__ X, const float* __restrict__ gamma,
                     u16* __restrict__ obf, float* __restrict__ of) {
  __shared__ float sred[256];
  int row = blockIdx.x, t = threadIdx.x;
  const float* xr = X + (size_t)row * DIM;
  float vals[4];
  float s = 0.f;
#pragma unroll
  for (int i = 0; i < 4; ++i) { vals[i] = xr[t + i * 256]; s += vals[i]; }
  sred[t] = s; __syncthreads();
  for (int k = 128; k > 0; k >>= 1) { if (t < k) sred[t] += sred[t + k]; __syncthreads(); }
  float mu = sred[0] * (1.0f / DIM);
  __syncthreads();
  float v = 0.f;
#pragma unroll
  for (int i = 0; i < 4; ++i) { float d = vals[i] - mu; v += d * d; }
  sred[t] = v; __syncthreads();
  for (int k = 128; k > 0; k >>= 1) { if (t < k) sred[t] += sred[t + k]; __syncthreads(); }
  float rstd = rsqrtf(sred[0] * (1.0f / DIM) + 1e-5f);
#pragma unroll
  for (int i = 0; i < 4; ++i) {
    int c = t + i * 256;
    float y = (vals[i] - mu) * rstd * gamma[c];
    if (obf) obf[(size_t)row * DIM + c] = f2bf(y);
    if (of)  of[(size_t)row * DIM + c]  = y;
  }
}

// --------------------------- generic bf16 WMMA GEMM ------------------------
// C[h] = A[h] (MxK, row-major bf16) x Bt[h]^T (Bt is NxK row-major bf16)
// wave = 32x64 tile (2 M-tiles share B-fragments), block = 128 thr (4 waves,
// 128 rows), grid = (M/128, N/64, heads)
__global__ void k_gemm(const u16* __restrict__ A, int lda, long long aHS,
                       const u16* __restrict__ Bt, int ldb, long long bHS,
                       float* __restrict__ Cf, u16* __restrict__ Cb,
                       u16* __restrict__ CbT,
                       int ldc, long long cHS, int ldcT, long long cTHS,
                       const float* __restrict__ bias, float scale, int K) {
  int h    = blockIdx.z;
  int lane = threadIdx.x & 31;
  int w    = threadIdx.x >> 5;
  int m0   = blockIdx.x * 128 + w * 32;
  int n0   = blockIdx.y * 64;
  const u16* Ah = A + (size_t)h * aHS;
  const u16* Bh = Bt + (size_t)h * bHS;

  v8f acc[2][4] = {};
  for (int k0 = 0; k0 < K; k0 += 32) {
    if (k0 + 64 < K) {   // prefetch next-next k-step A rows (global_prefetch_b8)
      __builtin_prefetch(Ah + (size_t)(m0 + (lane & 15)) * lda + k0 + 64, 0, 3);
      __builtin_prefetch(Ah + (size_t)(m0 + 16 + (lane & 15)) * lda + k0 + 64, 0, 3);
    }
    v16bf a0 = load_a_frag(Ah, lda, m0,      k0, lane);
    v16bf a1 = load_a_frag(Ah, lda, m0 + 16, k0, lane);
#pragma unroll
    for (int nt = 0; nt < 4; ++nt) {
      v16bf b = load_b_frag(Bh, ldb, n0 + nt * 16, k0, lane);
      acc[0][nt] = wmma_bf16(a0, b, acc[0][nt]);
      acc[1][nt] = wmma_bf16(a1, b, acc[1][nt]);
    }
  }
  float* Cfh = Cf ? Cf + (size_t)h * cHS : nullptr;
  u16*   Cbh = Cb ? Cb + (size_t)h * cHS : nullptr;
  u16*   CTh = CbT ? CbT + (size_t)h * cTHS : nullptr;
#pragma unroll
  for (int mt2 = 0; mt2 < 2; ++mt2) {
    int mb = m0 + mt2 * 16;
#pragma unroll
    for (int nt = 0; nt < 4; ++nt) {
#pragma unroll
      for (int r = 0; r < 8; ++r) {
        int row = mb + r + ((lane & 16) ? 8 : 0);
        int col = n0 + nt * 16 + (lane & 15);
        float z = acc[mt2][nt][r] * scale + (bias ? bias[col] : 0.0f);
        if (Cfh) Cfh[(size_t)row * ldc + col] = z;
        if (Cbh) Cbh[(size_t)row * ldc + col] = f2bf(z);
        if (CTh) CTh[(size_t)col * ldcT + row] = f2bf(z);
      }
    }
  }
}

// --------------------------- causal attention softmax ----------------------
// Per wave: 16-row strip of one head. Stores unnormalized p=exp(s-max) (bf16)
// and 1/rowsum. Strip is zero-padded to a 32-column boundary so later
// K=32 fragment loads never read undefined attn memory.
__global__ void k_attn(const u16* __restrict__ qbf, const u16* __restrict__ kbf,
                       u16* __restrict__ attn, float* __restrict__ rowinv) {
  int lane = threadIdx.x & 31;
  int w    = threadIdx.x >> 5;
  int mt   = blockIdx.x * 4 + w;
  int h    = blockIdx.y;
  int m0   = mt * 16;
  const u16* qh = qbf + h * DHEAD;
  const u16* kh = kbf + h * DHEAD;
  const float NEGF = -3.0e38f;

  v16bf aq0 = load_a_frag(qh, DIM, m0, 0, lane);
  v16bf aq1 = load_a_frag(qh, DIM, m0, 32, lane);

  v8f rmax;
#pragma unroll
  for (int r = 0; r < 8; ++r) rmax[r] = NEGF;

  // phase 1: row maxima
  for (int jt = 0; jt <= mt; ++jt) {
    if (jt < mt) {  // prefetch next K tile rows
      __builtin_prefetch(kh + (size_t)((jt + 1) * 16 + (lane & 15)) * DIM, 0, 3);
    }
    v8f acc = {};
    v16bf b0 = load_b_frag(kh, DIM, jt * 16, 0, lane);
    v16bf b1 = load_b_frag(kh, DIM, jt * 16, 32, lane);
    acc = wmma_bf16(aq0, b0, acc);
    acc = wmma_bf16(aq1, b1, acc);
#pragma unroll
    for (int r = 0; r < 8; ++r) {
      int row = m0 + r + ((lane & 16) ? 8 : 0);
      int col = jt * 16 + (lane & 15);
      float s = (col > row) ? NEGF : acc[r];
      rmax[r] = fmaxf(rmax[r], s);
    }
  }
#pragma unroll
  for (int r = 0; r < 8; ++r) {
    float m = rmax[r];
    for (int off = 1; off < 16; off <<= 1) m = fmaxf(m, __shfl_xor(m, off, 16));
    rmax[r] = m;
  }

  // phase 2: exp, store, row sums (pad to odd tile => 32-col multiple)
  v8f rsum = {};
  int jtEnd = mt | 1;
  for (int jt = 0; jt <= jtEnd; ++jt) {
    v8f acc = {};
    v16bf b0 = load_b_frag(kh, DIM, jt * 16, 0, lane);
    v16bf b1 = load_b_frag(kh, DIM, jt * 16, 32, lane);
    acc = wmma_bf16(aq0, b0, acc);
    acc = wmma_bf16(aq1, b1, acc);
#pragma unroll
    for (int r = 0; r < 8; ++r) {
      int row = m0 + r + ((lane & 16) ? 8 : 0);
      int col = jt * 16 + (lane & 15);
      float s = ((jt >= mt) && (col > row)) ? NEGF : acc[r];
      float p = __expf(s - rmax[r]);
      rsum[r] += p;
      attn[((size_t)(h * N_SEQ + row)) * N_SEQ + col] = f2bf(p);
    }
  }
#pragma unroll
  for (int r = 0; r < 8; ++r) {
    float s = rsum[r];
    for (int off = 1; off < 16; off <<= 1) s += __shfl_xor(s, off, 16);
    if ((lane & 15) == 0) {
      int row = m0 + r + ((lane & 16) ? 8 : 0);
      rowinv[h * N_SEQ + row] = 1.0f / s;
    }
  }
}

// --------------------------- adjacency degree ------------------------------
__global__ void k_deg(const u16* __restrict__ attn, const float* __restrict__ rowinv,
                      float* __restrict__ deg) {
  int lane = threadIdx.x & 31;
  int w    = threadIdx.x >> 5;
  int i    = blockIdx.x * 8 + w;
  int h    = blockIdx.y;
  const u16* rowp = attn + ((size_t)h * N_SEQ + i) * N_SEQ;
  float inv = rowinv[h * N_SEQ + i];
  float s = 0.f;
  for (int j = lane; j < i; j += 32) {
    float a = bf2f(rowp[j]) * inv;
    s += (a < 0.01f) ? 0.0f : ((a > 0.9f) ? 1.0f : a);
  }
  for (int off = 1; off < 32; off <<= 1) s += __shfl_xor(s, off, 32);
  if (lane == 0) deg[h * N_SEQ + i] = s + 1.0f;   // diagonal of adj is exactly 1
}

// --------------------------- attn/adjacency x Y matmul ---------------------
// mapped=0: C = softmax(attn) @ Y          (rowinv applied at epilogue)
// mapped=1: C = gelu( threshold(attn) @ Y / deg + skip ), also bf16 copy
// Yt: per-head transposed bf16 [head][64][N_SEQ].
// Wave handles the tile pair (2t, 2t+1): both strips share the zero-padded
// width (2t+2)*16 written by k_attn, and B-fragments are reused across tiles.
__global__ void k_adj(const u16* __restrict__ attn, const float* __restrict__ rowinv,
                      const float* __restrict__ deg, const u16* __restrict__ Yt,
                      const float* __restrict__ skip,
                      float* __restrict__ Cf, u16* __restrict__ Cb, int mapped) {
  int lane = threadIdx.x & 31;
  int w    = threadIdx.x >> 5;
  int mp   = blockIdx.x * 4 + w;        // tile-pair index
  int h    = blockIdx.y;
  int m0   = mp * 32;
  const u16* attnh = attn + (size_t)h * N_SEQ * N_SEQ;
  const u16* Yh    = Yt + (size_t)h * DHEAD * N_SEQ;
  int   arow0 = m0 + (lane & 15);
  int   arow1 = arow0 + 16;
  float ainv0 = rowinv[h * N_SEQ + arow0];
  float ainv1 = rowinv[h * N_SEQ + arow1];

  v8f acc[2][4] = {};
  int kmax = m0 + 32;                   // defined (zero-padded) strip width
  for (int k0 = 0; k0 < kmax; k0 += 32) {
    if (k0 + 64 < kmax) {   // prefetch next-next attn chunk
      __builtin_prefetch(attnh + (size_t)arow0 * N_SEQ + k0 + 64, 0, 3);
      __builtin_prefetch(attnh + (size_t)arow1 * N_SEQ + k0 + 64, 0, 3);
    }
    ABFrag f0, f1;
    int kb = k0 + ((lane & 16) ? 8 : 0);
    {
      const u16* p0 = attnh + (size_t)arow0 * N_SEQ + kb;
      const u16* p1 = attnh + (size_t)arow1 * N_SEQ + kb;
      f0.q[0] = *(const uint4*)(p0);
      f0.q[1] = *(const uint4*)(p0 + 16);
      f1.q[0] = *(const uint4*)(p1);
      f1.q[1] = *(const uint4*)(p1 + 16);
    }
    if (mapped) {
#pragma unroll
      for (int e = 0; e < 16; ++e) {
        int j = kb + (e < 8 ? e : 8 + e);
        float a0 = bf2f(f0.us[e]) * ainv0;
        float a1 = bf2f(f1.us[e]) * ainv1;
        float v0 = (j == arow0) ? 1.0f
                  : ((a0 < 0.01f) ? 0.0f : ((a0 > 0.9f) ? 1.0f : a0));
        float v1 = (j == arow1) ? 1.0f
                  : ((a1 < 0.01f) ? 0.0f : ((a1 > 0.9f) ? 1.0f : a1));
        f0.us[e] = f2bf(v0);
        f1.us[e] = f2bf(v1);
      }
    }
#pragma unroll
    for (int nt = 0; nt < 4; ++nt) {
      v16bf b = load_b_frag(Yh, N_SEQ, nt * 16, k0, lane);
      acc[0][nt] = wmma_bf16(f0.v, b, acc[0][nt]);
      acc[1][nt] = wmma_bf16(f1.v, b, acc[1][nt]);
    }
  }
#pragma unroll
  for (int mt2 = 0; mt2 < 2; ++mt2) {
    int mb = m0 + mt2 * 16;
#pragma unroll
    for (int nt = 0; nt < 4; ++nt) {
#pragma unroll
      for (int r = 0; r < 8; ++r) {
        int row = mb + r + ((lane & 16) ? 8 : 0);
        int col = h * DHEAD + nt * 16 + (lane & 15);
        float z = acc[mt2][nt][r];
        if (mapped) {
          z = z / deg[h * N_SEQ + row] + skip[(size_t)row * DIM + col];
          z = gelu_exact(z);
          Cf[(size_t)row * DIM + col] = z;
          Cb[(size_t)row * DIM + col] = f2bf(z);
        } else {
          Cf[(size_t)row * DIM + col] = z * rowinv[h * N_SEQ + row];
        }
      }
    }
  }
}

// --------------------------- merge heads (add + bf16) ----------------------
__global__ void k_add(const float* __restrict__ a, const float* __restrict__ b,
                      u16* __restrict__ o) {
  int idx = blockIdx.x * 256 + threadIdx.x;
  o[idx] = f2bf(a[idx] + b[idx]);
}

// ---------------------------------------------------------------------------
extern "C" void kernel_launch(void* const* d_in, const int* in_sizes, int n_in,
                              void* d_out, int out_size, void* d_ws, size_t ws_size,
                              hipStream_t stream) {
  (void)in_sizes; (void)n_in; (void)out_size; (void)ws_size;
  const float* x      = (const float*)d_in[0];
  const float* g_norm = (const float*)d_in[1];
  const float* Wq     = (const float*)d_in[2];
  const float* Wk     = (const float*)d_in[3];
  const float* Wv     = (const float*)d_in[4];
  const float* Wout   = (const float*)d_in[5];
  const float* g_out  = (const float*)d_in[6];
  const float* W0     = (const float*)d_in[7];
  const float* b0     = (const float*)d_in[8];
  const float* W1     = (const float*)d_in[9];
  const float* b1     = (const float*)d_in[10];
  const float* Wlin   = (const float*)d_in[11];
  const float* blin   = (const float*)d_in[12];
  float* out = (float*)d_out;

  char* p = (char*)d_ws;
  auto carve = [&](size_t bytes) -> void* {
    void* r = (void*)p;
    p += (bytes + 255) & ~(size_t)255;
    return r;
  };
  const size_t NS = N_SEQ, DD = DIM;
  u16*   xnbf   = (u16*)carve(NS * DD * 2);
  u16*   WqT    = (u16*)carve(DD * DD * 2);
  u16*   WkT    = (u16*)carve(DD * DD * 2);
  u16*   WvT    = (u16*)carve(DD * DD * 2);
  u16*   WoutT  = (u16*)carve(DD * DD * 2);
  u16*   W0T    = (u16*)carve(64 * 64 * 2);
  u16*   W1T    = (u16*)carve(64 * 64 * 2);
  u16*   WlinT  = (u16*)carve(64 * 64 * 2);
  u16*   qbf    = (u16*)carve(NS * DD * 2);
  u16*   kbf    = (u16*)carve(NS * DD * 2);
  u16*   vbf    = (u16*)carve(NS * DD * 2);
  u16*   vT     = (u16*)carve(DD * NS * 2);
  float* vf     = (float*)carve(NS * DD * 4);
  u16*   attn   = (u16*)carve((size_t)HEADS * NS * NS * 2);   // 128 MB
  float* rowinv = (float*)carve((size_t)HEADS * NS * 4);
  float* deg    = (float*)carve((size_t)HEADS * NS * 4);
  float* outat  = (float*)carve(NS * DD * 4);
  u16*   y0T    = (u16*)carve(DD * NS * 2);
  u16*   y1T    = (u16*)carve(DD * NS * 2);
  float* g1f    = (float*)carve(NS * DD * 4);
  u16*   g1b    = (u16*)carve(NS * DD * 2);
  float* g2f    = (float*)carve(NS * DD * 4);
  u16*   g2b    = (u16*)carve(NS * DD * 2);
  float* g3     = (float*)carve(NS * DD * 4);
  u16*   merged = (u16*)carve(NS * DD * 2);
  float* preLN  = (float*)carve(NS * DD * 4);

  const int ND = N_SEQ * DIM;
  dim3 blk128(128, 1, 1), blk256(256, 1, 1);

  // weights -> transposed bf16
  k_cvtT<<<(DIM * DIM + 255) / 256, blk256, 0, stream>>>(Wq,   WqT,   DIM, DIM);
  k_cvtT<<<(DIM * DIM + 255) / 256, blk256, 0, stream>>>(Wk,   WkT,   DIM, DIM);
  k_cvtT<<<(DIM * DIM + 255) / 256, blk256, 0, stream>>>(Wv,   WvT,   DIM, DIM);
  k_cvtT<<<(DIM * DIM + 255) / 256, blk256, 0, stream>>>(Wout, WoutT, DIM, DIM);
  k_cvtT<<<(64 * 64 + 255) / 256,   blk256, 0, stream>>>(W0,   W0T,   64, 64);
  k_cvtT<<<(64 * 64 + 255) / 256,   blk256, 0, stream>>>(W1,   W1T,   64, 64);
  k_cvtT<<<(64 * 64 + 255) / 256,   blk256, 0, stream>>>(Wlin, WlinT, 64, 64);

  // LN1 -> bf16
  k_ln<<<N_SEQ, blk256, 0, stream>>>(x, g_norm, xnbf, nullptr);

  dim3 gBig(N_SEQ / 128, DIM / 64, 1);
  // Q (scale folded), K, V (f32 + bf16 + transposed bf16)
  k_gemm<<<gBig, blk128, 0, stream>>>(xnbf, DIM, 0, WqT, DIM, 0,
      nullptr, qbf, nullptr, DIM, 0, 0, 0, nullptr, 0.125f, DIM);
  k_gemm<<<gBig, blk128, 0, stream>>>(xnbf, DIM, 0, WkT, DIM, 0,
      nullptr, kbf, nullptr, DIM, 0, 0, 0, nullptr, 1.0f, DIM);
  k_gemm<<<gBig, blk128, 0, stream>>>(xnbf, DIM, 0, WvT, DIM, 0,
      vf, vbf, vT, DIM, 0, N_SEQ, 0, nullptr, 1.0f, DIM);

  // causal softmax -> attn(bf16, unnormalized) + rowinv
  dim3 gAttn(N_SEQ / 64, HEADS, 1);
  k_attn<<<gAttn, blk128, 0, stream>>>(qbf, kbf, attn, rowinv);
  k_deg<<<dim3(N_SEQ / 8, HEADS, 1), blk256, 0, stream>>>(attn, rowinv, deg);

  // out_attn = attn @ v
  dim3 gAdj(N_SEQ / 128, HEADS, 1);
  k_adj<<<gAdj, blk128, 0, stream>>>(attn, rowinv, deg, vT, nullptr, outat, nullptr, 0);

  // GCN layer 0
  dim3 gSmall(N_SEQ / 128, 1, HEADS);
  k_gemm<<<gSmall, blk128, 0, stream>>>(vbf, DIM, DHEAD, W0T, 64, 0,
      nullptr, nullptr, y0T, 0, 0, N_SEQ, (long long)DHEAD * N_SEQ, b0, 1.0f, 64);
  k_adj<<<gAdj, blk128, 0, stream>>>(attn, rowinv, deg, y0T, vf, g1f, g1b, 1);

  // GCN layer 1
  k_gemm<<<gSmall, blk128, 0, stream>>>(g1b, DIM, DHEAD, W1T, 64, 0,
      nullptr, nullptr, y1T, 0, 0, N_SEQ, (long long)DHEAD * N_SEQ, b1, 1.0f, 64);
  k_adj<<<gAdj, blk128, 0, stream>>>(attn, rowinv, deg, y1T, g1f, g2f, g2b, 1);

  // final linear of GCN path
  k_gemm<<<gSmall, blk128, 0, stream>>>(g2b, DIM, DHEAD, WlinT, 64, 0,
      g3, nullptr, nullptr, DIM, DHEAD, 0, 0, blin, 1.0f, 64);

  // merge heads: out_attn + g3 -> bf16
  k_add<<<ND / 256, blk256, 0, stream>>>(outat, g3, merged);

  // output projection + LN2
  k_gemm<<<gBig, blk128, 0, stream>>>(merged, DIM, 0, WoutT, DIM, 0,
      preLN, nullptr, nullptr, DIM, 0, 0, 0, nullptr, 1.0f, DIM);
  k_ln<<<N_SEQ, blk256, 0, stream>>>(preLN, g_out, nullptr, out);
}
